// MultiLayerGAT_30709016167272
// MI455X (gfx1250) — matrix operand order repaired and checked by reading, not compiled, not run on hardware
//
#include <hip/hip_runtime.h>
#include <hip/hip_bf16.h>
#include <math.h>

// ---------------------------------------------------------------------------
// MultiLayerGAT for MI455X (gfx1250, wave32).
//  - GEMMs via v_wmma_f32_16x16x32_f16 (f32 accumulate), fused input-ReLU.
//  - Edge softmax/scatter: L2-resident gathers + f32 atomics (h fits in 192MB L2).
// ---------------------------------------------------------------------------

typedef __attribute__((ext_vector_type(16))) _Float16 v16h;
typedef __attribute__((ext_vector_type(8)))  float    v8f;

#define NEG_SLOPE 0.2f

// monotone float <-> uint key (valid for all finite floats and +-inf)
__device__ __forceinline__ unsigned f2key(float f) {
  unsigned b = __float_as_uint(f);
  return (b & 0x80000000u) ? ~b : (b | 0x80000000u);
}
__device__ __forceinline__ float key2f(unsigned k) {
  unsigned b = (k & 0x80000000u) ? (k & 0x7FFFFFFFu) : ~k;
  return __uint_as_float(b);
}

// Wt[n][k] = (f16) W[k][n]   (W is din x dout row-major)
__global__ void k_prep_wt(const float* __restrict__ W, _Float16* __restrict__ Wt,
                          int din, int dout) {
  int idx = blockIdx.x * blockDim.x + threadIdx.x;
  if (idx >= din * dout) return;
  int n = idx / din, k = idx - n * din;
  Wt[idx] = (_Float16)W[(long)k * dout + n];
}

// H[nrows,dout] = act(In[nrows,DIN]) @ W, with Wt = W^T in f16.
// One wave per 16-row strip; loops over dout/16 column tiles; K=DIN via
// DIN/32 v_wmma_f32_16x16x32_f16 per tile. EXEC is uniform per wave.
template <int DIN>
__global__ void __launch_bounds__(256) k_gemm_wmma(
    const float* __restrict__ In, const _Float16* __restrict__ Wt,
    float* __restrict__ H, int nrows, int dout, int relu_in) {
  constexpr int NK = DIN / 32;
  const int lane  = threadIdx.x & 31;
  const int strip = blockIdx.x * (blockDim.x >> 5) + (threadIdx.x >> 5);
  if (strip * 16 >= nrows) return;
  const int m  = lane & 15;   // row within tile (A/M), also col (B,D/N)
  const int hs = lane >> 4;   // half-select

  // A fragment layout (16-bit A 16x32): lanes 0-15 hold K = c*32 + {0..7, 16..23},
  // lanes 16-31 hold K = c*32 + {8..15, 24..31}; M = lane%16.
  const float* __restrict__ inrow = In + (long)(strip * 16 + m) * DIN;
  v16h A[NK];
#pragma unroll
  for (int c = 0; c < NK; ++c) {
    const int kb = c * 32 + hs * 8;
#pragma unroll
    for (int j = 0; j < 8; ++j) {
      float v0 = inrow[kb + j];
      float v1 = inrow[kb + 16 + j];
      if (relu_in) { v0 = fmaxf(v0, 0.0f); v1 = fmaxf(v1, 0.0f); }
      A[c][j]     = (_Float16)v0;
      A[c][j + 8] = (_Float16)v1;
    }
  }

  const int ntiles = dout >> 4;
  for (int t = 0; t < ntiles; ++t) {
    v8f acc = {};
    // B layout (16-bit B 32x16): lane n holds column n; lanes 0-15: K={0..7,16..23},
    // lanes 16-31: K={8..15,24..31} of Wt row (t*16+n), matching A.
    const _Float16* __restrict__ wt = Wt + (long)(t * 16 + m) * DIN + hs * 8;
#pragma unroll
    for (int c = 0; c < NK; ++c) {
      v16h B;
#pragma unroll
      for (int j = 0; j < 16; ++j) B[j] = wt[c * 32 + (j < 8 ? j : j + 8)];
      acc = __builtin_amdgcn_wmma_f32_16x16x32_f16(
          false, A[c], false, B, (short)0, acc, false, false);
    }
    // D layout: VGPR v -> row (v + hs*8), col = lane%16.
    float* __restrict__ hout =
        H + (long)(strip * 16 + hs * 8) * dout + t * 16 + m;
#pragma unroll
    for (int v = 0; v < 8; ++v) hout[(long)v * dout] = acc[v];
  }
}

// per-node attention dots: as_[i] = h_i . a_s ; ad_[i] = h_i . a_d (wave/node)
__global__ void __launch_bounds__(256) k_alpha(
    const float* __restrict__ h, const float* __restrict__ a_s,
    const float* __restrict__ a_d, float* __restrict__ as_,
    float* __restrict__ ad_, int N, int dout) {
  int node = blockIdx.x * (blockDim.x >> 5) + (threadIdx.x >> 5);
  if (node >= N) return;
  int lane = threadIdx.x & 31;
  const float* hr = h + (long)node * dout;
  float s = 0.f, d = 0.f;
  for (int c = lane; c < dout; c += 32) {
    float v = hr[c];
    s = fmaf(v, a_s[c], s);
    d = fmaf(v, a_d[c], d);
  }
  for (int off = 16; off > 0; off >>= 1) {
    s += __shfl_xor(s, off, 32);
    d += __shfl_xor(d, off, 32);
  }
  if (lane == 0) { as_[node] = s; ad_[node] = d; }
}

__global__ void k_init_node(unsigned* __restrict__ mkey,
                            float* __restrict__ denom, int N) {
  int i = blockIdx.x * blockDim.x + threadIdx.x;
  if (i >= N) return;
  mkey[i]  = f2key(-INFINITY);
  denom[i] = 0.0f;
}

__global__ void k_init_out(float* __restrict__ out, const float* __restrict__ b,
                           int N, int dout) {
  long idx = (long)blockIdx.x * blockDim.x + threadIdx.x;
  if (idx >= (long)N * dout) return;
  out[idx] = b[idx % dout];
}

// pass 1: logits + segment-max (ordered-uint atomicMax; tables are L2-resident)
__global__ void k_edge_logit_max(const int* __restrict__ src,
                                 const int* __restrict__ dst,
                                 const float* __restrict__ as_,
                                 const float* __restrict__ ad_,
                                 float* __restrict__ elog,
                                 unsigned* __restrict__ mkey, int E) {
  int e = blockIdx.x * blockDim.x + threadIdx.x;
  if (e >= E) return;
  float x = as_[src[e]] + ad_[dst[e]];
  float lg = x > 0.0f ? x : NEG_SLOPE * x;
  elog[e] = lg;
  atomicMax(&mkey[dst[e]], f2key(lg));
}

// pass 2: p = exp(e - max), denominator segment-sum
__global__ void k_edge_exp_sum(const int* __restrict__ dst,
                               const float* __restrict__ elog,
                               const unsigned* __restrict__ mkey,
                               float* __restrict__ pbuf,
                               float* __restrict__ denom, int E) {
  int e = blockIdx.x * blockDim.x + threadIdx.x;
  if (e >= E) return;
  int d = dst[e];
  float p = expf(elog[e] - key2f(mkey[d]));
  pbuf[e] = p;
  atomicAdd(&denom[d], p);
}

// pass 3: out[dst] += alpha * h[src]  (wave/edge, float4 gathers, f32 atomics)
__global__ void __launch_bounds__(256) k_scatter(
    const int* __restrict__ src, const int* __restrict__ dst,
    const float* __restrict__ pbuf, const float* __restrict__ denom,
    const float* __restrict__ h, float* __restrict__ out, int E, int dout) {
  int e = blockIdx.x * (blockDim.x >> 5) + (threadIdx.x >> 5);
  if (e >= E) return;
  int lane = threadIdx.x & 31;
  int s = src[e], d = dst[e];
  float alpha = pbuf[e] / (denom[d] + 1e-16f);
  const float4* hs = (const float4*)(h + (long)s * dout);
  float* od = out + (long)d * dout;
  for (int c4 = lane; c4 * 4 < dout; c4 += 32) {
    float4 v = hs[c4];
    atomicAdd(&od[c4 * 4 + 0], alpha * v.x);
    atomicAdd(&od[c4 * 4 + 1], alpha * v.y);
    atomicAdd(&od[c4 * 4 + 2], alpha * v.z);
    atomicAdd(&od[c4 * 4 + 3], alpha * v.w);
  }
}

// ---- graph-mean readout ----
__global__ void k_gzero(float* __restrict__ gsum, float* __restrict__ gcnt,
                        int G, int OUTD) {
  int i = blockIdx.x * blockDim.x + threadIdx.x;
  if (i < G * OUTD) gsum[i] = 0.0f;
  if (i < G) gcnt[i] = 0.0f;
}
__global__ void k_gcount(const int* __restrict__ batch,
                         float* __restrict__ gcnt, int N) {
  int i = blockIdx.x * blockDim.x + threadIdx.x;
  if (i >= N) return;
  atomicAdd(&gcnt[batch[i]], 1.0f);
}
__global__ void k_gsum(const float* __restrict__ h3,
                       const int* __restrict__ batch,
                       float* __restrict__ gsum, int N, int OUTD) {
  long idx = (long)blockIdx.x * blockDim.x + threadIdx.x;
  if (idx >= (long)N * OUTD) return;
  int i = (int)(idx / OUTD), c = (int)(idx - (long)i * OUTD);
  atomicAdd(&gsum[batch[i] * OUTD + c], h3[idx]);
}
__global__ void k_gfinal(const float* __restrict__ gsum,
                         const float* __restrict__ gcnt,
                         float* __restrict__ out, int G, int OUTD) {
  int idx = blockIdx.x * blockDim.x + threadIdx.x;
  if (idx >= G * OUTD) return;
  out[idx] = gsum[idx] / fmaxf(gcnt[idx / OUTD], 1.0f);
}

extern "C" void kernel_launch(void* const* d_in, const int* in_sizes, int n_in,
                              void* d_out, int out_size, void* d_ws,
                              size_t ws_size, hipStream_t stream) {
  (void)in_sizes; (void)n_in; (void)out_size; (void)ws_size;
  constexpr int N = 50000, E = 1600000, G = 64;
  constexpr int D_OUT = 64;

  const float* x     = (const float*)d_in[0];
  const int*   edge  = (const int*)d_in[1];
  const int*   batch = (const int*)d_in[2];
  const int*   src   = edge;
  const int*   dst   = edge + E;
  const float* Wm[3] = {(const float*)d_in[3], (const float*)d_in[7],  (const float*)d_in[11]};
  const float* As[3] = {(const float*)d_in[4], (const float*)d_in[8],  (const float*)d_in[12]};
  const float* Ad[3] = {(const float*)d_in[5], (const float*)d_in[9],  (const float*)d_in[13]};
  const float* Bs[3] = {(const float*)d_in[6], (const float*)d_in[10], (const float*)d_in[14]};

  // workspace carve-out (256B aligned)
  char* w = (char*)d_ws;
  size_t o = 0;
  auto take = [&](size_t bytes) -> void* {
    void* p = w + o;
    o += (bytes + 255) & ~(size_t)255;
    return p;
  };
  float* buf[3] = {(float*)take((size_t)N * 128 * 4),
                   (float*)take((size_t)N * 128 * 4),
                   (float*)take((size_t)N * 128 * 4)};
  float*    as_   = (float*)take((size_t)N * 4);
  float*    ad_   = (float*)take((size_t)N * 4);
  unsigned* mkey  = (unsigned*)take((size_t)N * 4);
  float*    denom = (float*)take((size_t)N * 4);
  float*    elog  = (float*)take((size_t)E * 4);
  float*    pbuf  = (float*)take((size_t)E * 4);
  float*    gsum  = (float*)take((size_t)G * D_OUT * 4);
  float*    gcnt  = (float*)take((size_t)G * 4);
  _Float16* wt    = (_Float16*)take((size_t)128 * 128 * 2);

  // buffer rotation: in -> h -> out per layer
  const float* lin[3]  = {x, buf[1], buf[0]};
  float*       lh[3]   = {buf[0], buf[2], buf[1]};
  float*       lout[3] = {buf[1], buf[0], buf[2]};
  const int    ldout[3] = {128, 128, 64};
  const int    lrelu[3] = {0, 1, 1};

  const int strips = (N + 15) / 16;  // 3125 (exact)
  for (int l = 0; l < 3; ++l) {
    const int dout = ldout[l];
    k_prep_wt<<<(128 * dout + 255) / 256, 256, 0, stream>>>(Wm[l], wt, 128, dout);
    k_gemm_wmma<128><<<(strips + 7) / 8, 256, 0, stream>>>(
        lin[l], wt, lh[l], N, dout, lrelu[l]);
    k_alpha<<<(N + 7) / 8, 256, 0, stream>>>(lh[l], As[l], Ad[l], as_, ad_, N, dout);
    k_init_node<<<(N + 255) / 256, 256, 0, stream>>>(mkey, denom, N);
    k_init_out<<<(int)(((long)N * dout + 255) / 256), 256, 0, stream>>>(
        lout[l], Bs[l], N, dout);
    k_edge_logit_max<<<(E + 255) / 256, 256, 0, stream>>>(src, dst, as_, ad_, elog, mkey, E);
    k_edge_exp_sum<<<(E + 255) / 256, 256, 0, stream>>>(dst, elog, mkey, pbuf, denom, E);
    k_scatter<<<(E + 7) / 8, 256, 0, stream>>>(src, dst, pbuf, denom, lh[l], lout[l], E, dout);
  }

  // graph-mean readout
  k_gzero<<<(G * D_OUT + 255) / 256, 256, 0, stream>>>(gsum, gcnt, G, D_OUT);
  k_gcount<<<(N + 255) / 256, 256, 0, stream>>>(batch, gcnt, N);
  k_gsum<<<(int)(((long)N * D_OUT + 255) / 256), 256, 0, stream>>>(
      lout[2], batch, gsum, N, D_OUT);
  k_gfinal<<<(G * D_OUT + 255) / 256, 256, 0, stream>>>(
      gsum, gcnt, (float*)d_out, G, D_OUT);
}
